// RoPEMultiHeadAttention_65094524339037
// MI455X (gfx1250) — matrix operand order repaired and checked by reading, not compiled, not run on hardware
//
#include <hip/hip_runtime.h>
#include <hip/hip_bf16.h>
#include <stdint.h>

typedef __attribute__((ext_vector_type(16))) __bf16 v16bf;
typedef __attribute__((ext_vector_type(8)))  __bf16 v8bf;
typedef __attribute__((ext_vector_type(4)))  __bf16 v4bf;
typedef __attribute__((ext_vector_type(8)))  float  v8f;

#define EMB   256
#define HEADS 8
#define DH    32
#define TSEQ  2048
#define BATCH 8
#define NQKV  768

// ---------------------------------------------------------------------------
// CDNA5 async global->LDS copy (ASYNCcnt-tracked, GVS addressing)
// ---------------------------------------------------------------------------
static __device__ __forceinline__ void async_copy_b128(uint32_t lds_addr,
                                                       const void* gbase,
                                                       uint32_t byte_off) {
    asm volatile("global_load_async_to_lds_b128 %0, %1, %2"
                 :
                 : "v"(lds_addr), "v"(byte_off), "s"((uint64_t)(uintptr_t)gbase)
                 : "memory");
}
static __device__ __forceinline__ void wait_async() {
    asm volatile("s_wait_asynccnt 0x0" ::: "memory");
}
static __device__ __forceinline__ uint32_t lds_off(const void* p) {
    return (uint32_t)(uintptr_t)p;   // low 32 bits of generic LDS addr = offset
}

// ---------------------------------------------------------------------------
// WMMA fragment loaders (layouts per CDNA5 ISA 7.12.2, wave32)
// ---------------------------------------------------------------------------
// A fragment: 16x32 bf16, source row-major [row][k], stride ldk.
static __device__ __forceinline__ v16bf load_a_frag(const __bf16* base, int ldk, int lane) {
    int row = lane & 15;
    int kb  = (lane >> 4) ? 8 : 0;
    const __bf16* p = base + row * ldk + kb;
    v8bf lo = *(const v8bf*)(p);        // K = kb .. kb+7
    v8bf hi = *(const v8bf*)(p + 16);   // K = kb+16 .. kb+23
    v16bf a;
#pragma unroll
    for (int i = 0; i < 8; i++) { a[i] = lo[i]; a[i + 8] = hi[i]; }
    return a;
}
// B fragment: 32x16 bf16 (KxN), source stored as [n][k] row-major, stride ldk.
static __device__ __forceinline__ v16bf load_b_frag(const __bf16* base, int ldk, int lane) {
    int n  = lane & 15;
    int kb = (lane >> 4) ? 8 : 0;
    const __bf16* p = base + n * ldk + kb;
    v8bf lo = *(const v8bf*)(p);
    v8bf hi = *(const v8bf*)(p + 16);
    v16bf b;
#pragma unroll
    for (int i = 0; i < 8; i++) { b[i] = lo[i]; b[i + 8] = hi[i]; }
    return b;
}
static __device__ __forceinline__ v8f wmma_bf16(v16bf a, v16bf b, v8f c) {
    return __builtin_amdgcn_wmma_f32_16x16x32_bf16(
        false, a, false, b, (short)0, c, false, false);
}

// ---------------------------------------------------------------------------
// Kernel 1: qkv = x @ Wqkv + bqkv, fused RoPE on q/k, scatter to q/k/vT bufs
// Grid: (256, 12). Block: 128 threads (4 waves).
// ---------------------------------------------------------------------------
__global__ __launch_bounds__(128) void qkv_rope_kernel(
    const float* __restrict__ x, const float* __restrict__ Wqkv,
    const float* __restrict__ bqkv,
    __bf16* __restrict__ qbuf, __bf16* __restrict__ kbuf, __bf16* __restrict__ vbuf)
{
    __shared__ __attribute__((aligned(16))) __bf16 As[64 * 32];   // [m][k]
    __shared__ __attribute__((aligned(16))) __bf16 Bs[64 * 32];   // [n][k]

    const int tid  = threadIdx.x;
    const int lane = tid & 31;
    const int wave = tid >> 5;
    const int m0 = blockIdx.x * 64;
    const int n0 = blockIdx.y * 64;

    v8f zero = {};
    v8f acc[4] = {zero, zero, zero, zero};

    for (int k0 = 0; k0 < EMB; k0 += 32) {
        __syncthreads();
        // A tile: 64x32 f32 -> bf16, 512 float4 loads / 128 thr = 4 each
#pragma unroll
        for (int i = 0; i < 4; i++) {
            int lin = tid + i * 128;                // 0..511
            int r = lin >> 3, c4 = (lin & 7) << 2;  // 8 float4 per row
            float4 v = *(const float4*)&x[(size_t)(m0 + r) * EMB + k0 + c4];
            v4bf t;
            t[0] = (__bf16)v.x; t[1] = (__bf16)v.y;
            t[2] = (__bf16)v.z; t[3] = (__bf16)v.w;
            *(v4bf*)&As[r * 32 + c4] = t;
        }
        // B tile (transposed store): float4 along n, 4 scalar LDS stores
#pragma unroll
        for (int i = 0; i < 4; i++) {
            int lin = tid + i * 128;                 // 0..511
            int k = lin >> 4, n4 = (lin & 15) << 2;  // 16 n-groups per k
            float4 v = *(const float4*)&Wqkv[(size_t)(k0 + k) * NQKV + n0 + n4];
            Bs[(n4 + 0) * 32 + k] = (__bf16)v.x;
            Bs[(n4 + 1) * 32 + k] = (__bf16)v.y;
            Bs[(n4 + 2) * 32 + k] = (__bf16)v.z;
            Bs[(n4 + 3) * 32 + k] = (__bf16)v.w;
        }
        __syncthreads();
        v16bf af = load_a_frag(&As[wave * 16 * 32], 32, lane);
#pragma unroll
        for (int t = 0; t < 4; t++) {
            v16bf bf = load_b_frag(&Bs[t * 16 * 32], 32, lane);
            acc[t] = wmma_bf16(af, bf, acc[t]);
        }
    }

    const float inv_sqrt_dh = 0.17677669529663687f;      // 1/sqrt(32)
    const float neg_ln_1e4_over_dh = -0.287823136624256f; // -ln(10000)/32
    const int rowofs = (lane >> 4) ? 8 : 0;
#pragma unroll
    for (int t = 0; t < 4; t++) {
        // per-tile invariants: column -> (part, h, d) and RoPE frequency
        int gn   = n0 + t * 16 + (lane & 15);
        int part = gn >> 8;       // 0=q 1=k 2=v (uniform within a 16-wide tile)
        int rem  = gn & 255;
        int h    = rem >> 5;
        int d    = rem & 31;
        float freq = __expf((float)(d & ~1) * neg_ln_1e4_over_dh);
        float bias = bqkv[gn];
#pragma unroll
        for (int r = 0; r < 8; r++) {
            int gm  = m0 + wave * 16 + r + rowofs;
            int b   = gm >> 11;
            int tok = gm & 2047;
            float c = acc[t][r] + bias;

            float partner = __shfl_xor(c, 1, 32);   // RoPE pair = adjacent lane
            if (part < 2) {
                float ang = (float)tok * freq;
                float sn, cs;
                __sincosf(ang, &sn, &cs);
                c = (d & 1) ? (c * cs + partner * sn) : (c * cs - partner * sn);
            }
            size_t bh = (size_t)(b * HEADS + h);
            if (part == 0) {
                qbuf[(bh * TSEQ + tok) * DH + d] = (__bf16)(c * inv_sqrt_dh);
            } else if (part == 1) {
                kbuf[(bh * TSEQ + tok) * DH + d] = (__bf16)c;
            } else {
                vbuf[(bh * DH + d) * TSEQ + tok] = (__bf16)c;  // V transposed
            }
        }
    }
}

// ---------------------------------------------------------------------------
// Kernel 2: causal flash attention. Each wave owns 16 queries; WG = 64 queries.
// K/V chunks staged with GLOBAL_LOAD_ASYNC_TO_LDS_B128 (pure bf16 byte copy).
// Grid: (B*H, T/64) = (64, 32). Block: 128 threads.
// ---------------------------------------------------------------------------
__global__ __launch_bounds__(128) void attn_kernel(
    const __bf16* __restrict__ qbuf, const __bf16* __restrict__ kbuf,
    const __bf16* __restrict__ vbuf, __bf16* __restrict__ att)
{
    __shared__ __attribute__((aligned(16))) __bf16 Kt[32 * 32];    // [key][d]
    __shared__ __attribute__((aligned(16))) __bf16 Vt[32 * 32];    // [d][key]
    __shared__ __attribute__((aligned(16))) __bf16 Pl[4][16 * 32]; // per-wave P

    const int tid  = threadIdx.x;
    const int lane = tid & 31;
    const int wave = tid >> 5;
    const int bh   = blockIdx.x;
    const int b    = bh >> 3;
    const int h    = bh & 7;
    const int q0   = blockIdx.y * 64 + wave * 16;
    const int rowofs = (lane >> 4) ? 8 : 0;

    v16bf qfrag = load_a_frag(qbuf + ((size_t)bh * TSEQ + q0) * DH, DH, lane);

    v8f zero = {};
    v8f o0 = zero, o1 = zero;
    float m[8], l[8];
#pragma unroll
    for (int r = 0; r < 8; r++) { m[r] = -__builtin_inff(); l[r] = 0.f; }

    // per-thread async-copy geometry: 16 bytes (8 bf16) each
    const int crow = tid >> 2;          // 0..31
    const int c8   = (tid & 3) << 3;    // 0,8,16,24
    const uint32_t kt_lds = lds_off(&Kt[0]) + tid * 16;
    const uint32_t vt_lds = lds_off(&Vt[0]) + tid * 16;

    const int kmax = blockIdx.y * 64 + 64;
    for (int jj = 0; jj < kmax; jj += 32) {
        __syncthreads();
        // Kt[key][d] <- kbuf[bh][jj+key][d]   (one B128 per thread)
        async_copy_b128(kt_lds, kbuf,
            (uint32_t)((((bh * TSEQ) + jj + crow) * DH + c8) * 2));
        // Vt[d][key] <- vbuf[bh][d][jj+key]
        async_copy_b128(vt_lds, vbuf,
            (uint32_t)((((bh * DH) + crow) * TSEQ + jj + c8) * 2));
        if (jj + 32 < kmax) {  // prefetch next chunk into L2
            __builtin_prefetch(&kbuf[(((size_t)bh * TSEQ) + jj + 32 + crow) * DH], 0, 0);
            __builtin_prefetch(&vbuf[(((size_t)bh * DH) + crow) * TSEQ + jj + 32], 0, 0);
        }
        wait_async();
        __syncthreads();

        v8f s0 = wmma_bf16(qfrag, load_b_frag(&Kt[0],       32, lane), zero);
        v8f s1 = wmma_bf16(qfrag, load_b_frag(&Kt[16 * 32], 32, lane), zero);

        int col0 = jj + (lane & 15);
        int col1 = col0 + 16;
        float pv0[8], pv1[8];
#pragma unroll
        for (int r = 0; r < 8; r++) {
            int qrow = q0 + r + rowofs;
            float a  = (col0 <= qrow) ? s0[r] : -__builtin_inff();
            float bb = (col1 <= qrow) ? s1[r] : -__builtin_inff();
            float mx = fmaxf(a, bb);
#pragma unroll
            for (int off = 1; off < 16; off <<= 1)
                mx = fmaxf(mx, __shfl_xor(mx, off, 32));
            float mn = fmaxf(m[r], mx);
            float alpha = (m[r] == -__builtin_inff()) ? 0.f : __expf(m[r] - mn);
            float p0 = __expf(a - mn);
            float p1 = __expf(bb - mn);
            float ps = p0 + p1;
#pragma unroll
            for (int off = 1; off < 16; off <<= 1)
                ps += __shfl_xor(ps, off, 32);
            l[r] = l[r] * alpha + ps;
            m[r] = mn;
            o0[r] *= alpha;
            o1[r] *= alpha;
            pv0[r] = p0;
            pv1[r] = p1;
        }

        // transpose P (C-layout -> A-frag layout) through per-wave LDS
        __bf16* P = &Pl[wave][0];
#pragma unroll
        for (int r = 0; r < 8; r++) {
            P[(r + rowofs) * 32 + (lane & 15)]      = (__bf16)pv0[r];
            P[(r + rowofs) * 32 + 16 + (lane & 15)] = (__bf16)pv1[r];
        }
        v16bf pfrag = load_a_frag(P, 32, lane);
        o0 = wmma_bf16(pfrag, load_b_frag(&Vt[0],       32, lane), o0);
        o1 = wmma_bf16(pfrag, load_b_frag(&Vt[16 * 32], 32, lane), o1);
    }

#pragma unroll
    for (int r = 0; r < 8; r++) {
        float invl = 1.f / l[r];
        int tok = q0 + r + rowofs;
        size_t base = ((size_t)b * TSEQ + tok) * EMB + h * DH;
        att[base + (lane & 15)]      = (__bf16)(o0[r] * invl);
        att[base + 16 + (lane & 15)] = (__bf16)(o1[r] * invl);
    }
}

// ---------------------------------------------------------------------------
// Kernel 3: out = att @ Wout + bout. A-tiles (bf16) staged via async B128.
// Grid: (256, 4). Block: 128 threads.
// ---------------------------------------------------------------------------
__global__ __launch_bounds__(128) void out_proj_kernel(
    const __bf16* __restrict__ att, const float* __restrict__ Wout,
    const float* __restrict__ bout, float* __restrict__ out)
{
    __shared__ __attribute__((aligned(16))) __bf16 As[64 * 32];
    __shared__ __attribute__((aligned(16))) __bf16 Bs[64 * 32];

    const int tid  = threadIdx.x;
    const int lane = tid & 31;
    const int wave = tid >> 5;
    const int m0 = blockIdx.x * 64;
    const int n0 = blockIdx.y * 64;
    const uint32_t as_base = lds_off(&As[0]);

    v8f zero = {};
    v8f acc[4] = {zero, zero, zero, zero};

    for (int k0 = 0; k0 < EMB; k0 += 32) {
        __syncthreads();
        // A tile: pure bf16 copy -> async to LDS (2 x B128 per thread)
#pragma unroll
        for (int i = 0; i < 2; i++) {
            int lin = tid + i * 128;                // 0..255 chunks of 8 elems
            int r = lin >> 2, c8 = (lin & 3) << 3;  // row stride 32 elems
            async_copy_b128(as_base + lin * 16, att,
                (uint32_t)((((m0 + r) * EMB) + k0 + c8) * 2));
        }
        // B tile: f32 -> bf16, transposed store
#pragma unroll
        for (int i = 0; i < 4; i++) {
            int lin = tid + i * 128;
            int k = lin >> 4, n4 = (lin & 15) << 2;
            float4 v = *(const float4*)&Wout[(size_t)(k0 + k) * EMB + n0 + n4];
            Bs[(n4 + 0) * 32 + k] = (__bf16)v.x;
            Bs[(n4 + 1) * 32 + k] = (__bf16)v.y;
            Bs[(n4 + 2) * 32 + k] = (__bf16)v.z;
            Bs[(n4 + 3) * 32 + k] = (__bf16)v.w;
        }
        wait_async();
        __syncthreads();
        v16bf af = load_a_frag(&As[wave * 16 * 32], 32, lane);
#pragma unroll
        for (int t = 0; t < 4; t++) {
            v16bf bf = load_b_frag(&Bs[t * 16 * 32], 32, lane);
            acc[t] = wmma_bf16(af, bf, acc[t]);
        }
    }

    const int rowofs = (lane >> 4) ? 8 : 0;
#pragma unroll
    for (int t = 0; t < 4; t++) {
#pragma unroll
        for (int r = 0; r < 8; r++) {
            int gm = m0 + wave * 16 + r + rowofs;
            int gn = n0 + t * 16 + (lane & 15);
            out[(size_t)gm * EMB + gn] = acc[t][r] + bout[gn];
        }
    }
}

// ---------------------------------------------------------------------------
extern "C" void kernel_launch(void* const* d_in, const int* in_sizes, int n_in,
                              void* d_out, int out_size, void* d_ws, size_t ws_size,
                              hipStream_t stream) {
    const float* x    = (const float*)d_in[0];
    const float* Wqkv = (const float*)d_in[1];
    const float* bqkv = (const float*)d_in[2];
    const float* Wout = (const float*)d_in[3];
    const float* bout = (const float*)d_in[4];
    float* out = (float*)d_out;

    char* ws = (char*)d_ws;
    const size_t seg = (size_t)BATCH * HEADS * TSEQ * DH * sizeof(__bf16); // 8 MB
    __bf16* qbuf = (__bf16*)(ws + 0 * seg);
    __bf16* kbuf = (__bf16*)(ws + 1 * seg);
    __bf16* vbuf = (__bf16*)(ws + 2 * seg);
    __bf16* attb = (__bf16*)(ws + 3 * seg);

    qkv_rope_kernel<<<dim3(256, 12), 128, 0, stream>>>(x, Wqkv, bqkv, qbuf, kbuf, vbuf);
    attn_kernel<<<dim3(64, 32), 128, 0, stream>>>(qbuf, kbuf, vbuf, attb);
    out_proj_kernel<<<dim3(256, 4), 128, 0, stream>>>(attb, Wout, bout, out);
}
